// LiteMLA_74594991997231
// MI455X (gfx1250) — matrix operand order repaired and checked by reading, not compiled, not run on hardware
//
#include <hip/hip_runtime.h>
#include <hip/hip_bf16.h>

typedef __bf16 bf16;
typedef __attribute__((ext_vector_type(8)))  __bf16 v8bf_t;
typedef __attribute__((ext_vector_type(16))) __bf16 v16bf_t;
typedef __attribute__((ext_vector_type(8)))  float  v8f;

union V16 { v16bf_t v; v8bf_t h[2]; };

// ---------------------------------------------------------------------------
// 1) Transpose-convert hidden (B,512,64,64) f32 NCHW -> Xt[b][p][c] bf16
// ---------------------------------------------------------------------------
__global__ __launch_bounds__(256) void transpose_convert(
    const float* __restrict__ x, bf16* __restrict__ xt)
{
    // block = (32,8); grid = b(4) * ctile(16) * ptile(128)
    int bid = blockIdx.x;
    int pt = bid % 128;
    int ct = (bid / 128) % 16;
    int b  = bid / (128 * 16);
    __shared__ float tile[32][33];
    int p0 = pt * 32, c0 = ct * 32;
    #pragma unroll
    for (int i = 0; i < 4; ++i) {
        int c = c0 + threadIdx.y + i * 8;
        tile[threadIdx.y + i * 8][threadIdx.x] =
            x[((size_t)b * 512 + c) * 4096 + p0 + threadIdx.x];
    }
    __syncthreads();
    #pragma unroll
    for (int i = 0; i < 4; ++i) {
        int p = p0 + threadIdx.y + i * 8;
        xt[((size_t)b * 4096 + p) * 512 + c0 + threadIdx.x] =
            (bf16)tile[threadIdx.x][threadIdx.y + i * 8];
    }
}

// ---------------------------------------------------------------------------
// 2) Weight packing + BN prep
// ---------------------------------------------------------------------------
__global__ __launch_bounds__(256) void pack_wqkv(
    const float* __restrict__ wq, const float* __restrict__ wk,
    const float* __restrict__ wv, bf16* __restrict__ w)
{
    int i = blockIdx.x * 256 + threadIdx.x;  // 0 .. 1536*512-1
    int r = i >> 9;
    float v = (r < 512) ? wq[i] : (r < 1024 ? wk[i - 512 * 512] : wv[i - 1024 * 512]);
    w[i] = (bf16)v;
}

__global__ __launch_bounds__(256) void pack_proj(
    const float* __restrict__ pj, bf16* __restrict__ w)
{
    int i = blockIdx.x * 256 + threadIdx.x;  // 0 .. 512*1024-1
    w[i] = (bf16)pj[i];
}

__global__ __launch_bounds__(256) void bn_prep(
    const float* __restrict__ g, const float* __restrict__ be,
    const float* __restrict__ mn, const float* __restrict__ vr,
    float* __restrict__ inv, float* __restrict__ sh)
{
    int i = blockIdx.x * 256 + threadIdx.x;
    if (i < 512) {
        float iv = g[i] * rsqrtf(vr[i] + 1e-5f);
        inv[i] = iv;
        sh[i]  = be[i] - mn[i] * iv;
    }
}

// ---------------------------------------------------------------------------
// 3) WMMA GEMM, 4x4 register-blocked: each wave computes a 64x64 output
//    block (16 accumulators). Per 32-k step: 8 A loads + 8 B loads feed
//    16 v_wmma_f32_16x16x32_bf16 (load:wmma = 1:1).
//    A: lane = row m (K chunks of 8 at +0/+16); B: lane = col n (K chunks
//    of 8 at +0/+16). Optional fused BN+residual epilogue.
// ---------------------------------------------------------------------------
template <int KDIM, bool EPI>
__global__ __launch_bounds__(128, 1) void gemm_wmma(
    const bf16* __restrict__ W, const bf16* __restrict__ Xt,
    float* __restrict__ out, int Mblk,
    const float* __restrict__ bninv, const float* __restrict__ bnshift,
    const float* __restrict__ resid)
{
    const int lane = threadIdx.x & 31;
    const int wave = threadIdx.x >> 5;
    const int wid  = blockIdx.x * 4 + wave;
    const int mb   = wid % Mblk;            // 64-row block
    const int nb   = wid / Mblk;            // 64-col block (global)
    const int b    = nb >> 6;               // 64 n-blocks per batch
    const int n0   = (nb & 63) << 6;
    const int m0   = mb << 6;
    const int M    = Mblk << 6;

    const int l15 = lane & 15;
    const int lhi = lane >> 4;

    const bf16* arow = W  + (size_t)(m0 + l15) * KDIM + lhi * 8;
    const bf16* brow = Xt + ((size_t)b * 4096 + (n0 + l15)) * KDIM + lhi * 8;
    const size_t astep = (size_t)16 * KDIM;   // between 16-row A tiles
    const size_t bstep = (size_t)16 * KDIM;   // between 16-col B tiles

    v8f acc[4][4];
    #pragma unroll
    for (int i = 0; i < 4; ++i)
        #pragma unroll
        for (int j = 0; j < 4; ++j) acc[i][j] = (v8f){};

    #pragma unroll 1
    for (int k0 = 0; k0 < KDIM; k0 += 32) {
        V16 a[4], bm[4];
        #pragma unroll
        for (int i = 0; i < 4; ++i) {
            a[i].h[0] = *(const v8bf_t*)(arow + i * astep + k0);
            a[i].h[1] = *(const v8bf_t*)(arow + i * astep + k0 + 16);
        }
        #pragma unroll
        for (int j = 0; j < 4; ++j) {
            bm[j].h[0] = *(const v8bf_t*)(brow + j * bstep + k0);
            bm[j].h[1] = *(const v8bf_t*)(brow + j * bstep + k0 + 16);
        }
        __builtin_prefetch(arow + k0 + 32, 0, 0);
        __builtin_prefetch(brow + k0 + 32, 0, 0);
        #pragma unroll
        for (int j = 0; j < 4; ++j)
            #pragma unroll
            for (int i = 0; i < 4; ++i)
                acc[i][j] = __builtin_amdgcn_wmma_f32_16x16x32_bf16(
                    false, a[i].v, false, bm[j].v, (short)0, acc[i][j],
                    false, false);
    }

    // C layout: lane l, reg r -> (m = r + 8*(l>>4), n = l&15) within a tile
    #pragma unroll
    for (int i = 0; i < 4; ++i) {
        #pragma unroll
        for (int r = 0; r < 8; ++r) {
            int m = m0 + i * 16 + r + 8 * lhi;
            float sc = 0.f, sh = 0.f;
            if (EPI) { sc = bninv[m]; sh = bnshift[m]; }
            #pragma unroll
            for (int j = 0; j < 4; ++j) {
                int n = n0 + j * 16 + l15;
                size_t o = ((size_t)b * M + m) * 4096 + n;
                float v = acc[i][j][r];
                if (EPI) v = v * sc + sh + resid[o];
                out[o] = v;
            }
        }
    }
}

// ---------------------------------------------------------------------------
// 4) Depthwise 5x5 SAME conv (1536 channels), fp32
// ---------------------------------------------------------------------------
__global__ __launch_bounds__(256) void dwconv(
    const float* __restrict__ qkv, const float* __restrict__ kern,
    float* __restrict__ outp)
{
    int idx = blockIdx.x * 256 + threadIdx.x;   // B*1536*4096
    int p = idx & 4095;
    int c = (idx >> 12) % 1536;
    int b = idx / (4096 * 1536);
    int y = p >> 6, x = p & 63;
    const float* src = qkv + ((size_t)b * 1536 + c) * 4096;
    const float* kc  = kern + c * 25;
    float acc = 0.f;
    #pragma unroll
    for (int ky = 0; ky < 5; ++ky) {
        int iy = y + ky - 2;
        if ((unsigned)iy < 64u) {
            #pragma unroll
            for (int kx = 0; kx < 5; ++kx) {
                int ix = x + kx - 2;
                if ((unsigned)ix < 64u) acc += src[iy * 64 + ix] * kc[ky * 5 + kx];
            }
        }
    }
    outp[idx] = acc;
}

// ---------------------------------------------------------------------------
// 5) Grouped pointwise conv: 192 groups of 8->8 channels
// ---------------------------------------------------------------------------
__global__ __launch_bounds__(256) void pwconv(
    const float* __restrict__ dw, const float* __restrict__ kern,
    float* __restrict__ outp)
{
    int idx = blockIdx.x * 256 + threadIdx.x;   // B*1536*4096
    int p  = idx & 4095;
    int oc = (idx >> 12) % 1536;
    int b  = idx / (4096 * 1536);
    int g  = oc >> 3;
    const float* src = dw + ((size_t)b * 1536 + g * 8) * 4096 + p;
    const float* kc  = kern + oc * 8;
    float acc = 0.f;
    #pragma unroll
    for (int ic = 0; ic < 8; ++ic) acc += src[(size_t)ic * 4096] * kc[ic];
    outp[idx] = acc;
}

// ---------------------------------------------------------------------------
// 6) Linear attention: one workgroup per (batch, head).
//    Pass A: scores[9][8] via LDS f32 atomics; Pass B: normalize and write
//    bf16 transposed as attn_t[b][p][1024] for the projection GEMM.
// ---------------------------------------------------------------------------
__global__ __launch_bounds__(256) void attn_kernel(
    const float* __restrict__ qkv, const float* __restrict__ pw,
    bf16* __restrict__ attn_t)
{
    const int h = blockIdx.x & 127;
    const int b = blockIdx.x >> 7;
    const int base = h * 24;                 // channel base within 3072
    __shared__ float s[72];
    if (threadIdx.x < 72) s[threadIdx.x] = 0.f;
    __syncthreads();

    auto chptr = [&](int ch) -> const float* {
        return (ch < 1536) ? (qkv + ((size_t)b * 1536 + ch) * 4096)
                           : (pw  + ((size_t)b * 1536 + (ch - 1536)) * 4096);
    };
    const float *qp[8], *kp[8], *vp[8];
    #pragma unroll
    for (int e = 0; e < 8; ++e) {
        qp[e] = chptr(base + e);
        kp[e] = chptr(base + 8 + e);
        vp[e] = chptr(base + 16 + e);
    }

    float acc[72];
    #pragma unroll
    for (int i = 0; i < 72; ++i) acc[i] = 0.f;

    for (int n = threadIdx.x; n < 4096; n += 256) {
        float kk[8], vv[8];
        #pragma unroll
        for (int e = 0; e < 8; ++e) kk[e] = fmaxf(kp[e][n], 0.f);
        #pragma unroll
        for (int d = 0; d < 8; ++d) vv[d] = vp[d][n];
        #pragma unroll
        for (int d = 0; d < 8; ++d)
            #pragma unroll
            for (int e = 0; e < 8; ++e) acc[d * 8 + e] += vv[d] * kk[e];
        #pragma unroll
        for (int e = 0; e < 8; ++e) acc[64 + e] += kk[e];   // v_pad row = 1
    }
    #pragma unroll
    for (int i = 0; i < 72; ++i) atomicAdd(&s[i], acc[i]);  // ds_add_f32
    __syncthreads();

    float sc[72];
    #pragma unroll
    for (int i = 0; i < 72; ++i) sc[i] = s[i];

    for (int n = threadIdx.x; n < 4096; n += 256) {
        float qq[8];
        #pragma unroll
        for (int e = 0; e < 8; ++e) qq[e] = fmaxf(qp[e][n], 0.f);
        float o[9];
        #pragma unroll
        for (int d = 0; d < 9; ++d) {
            float t = 0.f;
            #pragma unroll
            for (int e = 0; e < 8; ++e) t += sc[d * 8 + e] * qq[e];
            o[d] = t;
        }
        float rinv = 1.f / (o[8] + 1e-15f);
        v8bf_t outv;
        #pragma unroll
        for (int d = 0; d < 8; ++d) outv[d] = (bf16)(o[d] * rinv);
        *(v8bf_t*)(attn_t + ((size_t)b * 4096 + n) * 1024 + h * 8) = outv;
    }
}

// ---------------------------------------------------------------------------
// Host launcher
// ---------------------------------------------------------------------------
extern "C" void kernel_launch(void* const* d_in, const int* in_sizes, int n_in,
                              void* d_out, int out_size, void* d_ws, size_t ws_size,
                              hipStream_t stream)
{
    const float* hidden = (const float*)d_in[0];
    const float* wq     = (const float*)d_in[1];
    const float* wk     = (const float*)d_in[2];
    const float* wv     = (const float*)d_in[3];
    const float* dwk    = (const float*)d_in[4];
    const float* pwk    = (const float*)d_in[5];
    const float* projw  = (const float*)d_in[6];
    const float* gamma  = (const float*)d_in[7];
    const float* beta   = (const float*)d_in[8];
    const float* mean   = (const float*)d_in[9];
    const float* var    = (const float*)d_in[10];
    float* out = (float*)d_out;

    // workspace layout (bytes, 256-aligned); attn_t aliases the dead dw buffer
    char* ws = (char*)d_ws;
    float* qkvb    = (float*)(ws);                         // 100,663,296
    float* dwb     = (float*)(ws + 100663296u);            // 100,663,296
    float* pwb     = (float*)(ws + 201326592u);            // 100,663,296
    bf16*  attn_t  = (bf16*) (ws + 100663296u);            // alias dwb (33.5 MB)
    bf16*  xt      = (bf16*) (ws + 301989888u);            //  16,777,216
    bf16*  wqkv    = (bf16*) (ws + 318767104u);            //   1,572,864
    bf16*  wproj   = (bf16*) (ws + 320339968u);            //   1,048,576
    float* bninv   = (float*)(ws + 321388544u);            //       2,048
    float* bnshift = (float*)(ws + 321390592u);            //       2,048

    // Stage 1: layout/precision prep
    transpose_convert<<<4 * 16 * 128, dim3(32, 8), 0, stream>>>(hidden, xt);
    pack_wqkv<<<(1536 * 512) / 256, 256, 0, stream>>>(wq, wk, wv, wqkv);
    pack_proj<<<(512 * 1024) / 256, 256, 0, stream>>>(projw, wproj);
    bn_prep<<<2, 256, 0, stream>>>(gamma, beta, mean, var, bninv, bnshift);

    // Stage 2: QKV projection GEMM (bf16 WMMA, f32 acc)
    // 64x64 blocks: (1536/64)=24 m-blocks * (4*4096/64)=256 n-blocks = 6144 waves
    gemm_wmma<512, false><<<6144 / 4, 128, 0, stream>>>(
        wqkv, xt, qkvb, 24, nullptr, nullptr, nullptr);

    // Stage 3/4: depthwise 5x5 then grouped pointwise
    dwconv<<<(4 * 1536 * 4096) / 256, 256, 0, stream>>>(qkvb, dwk, dwb);
    pwconv<<<(4 * 1536 * 4096) / 256, 256, 0, stream>>>(dwb, pwk, pwb);

    // Stage 5: linear attention per (batch, head); writes bf16 attn_t[b][p][c]
    attn_kernel<<<4 * 128, 256, 0, stream>>>(qkvb, pwb, attn_t);

    // Stage 6: projection GEMM + fused BN + residual -> d_out
    // (512/64)=8 m-blocks * 256 n-blocks = 2048 waves
    gemm_wmma<1024, true><<<2048 / 4, 128, 0, stream>>>(
        wproj, attn_t, out, 8, bninv, bnshift, hidden);
}
